// HeteroGAT_80169859547982
// MI455X (gfx1250) — matrix operand order repaired and checked by reading, not compiled, not run on hardware
//
#include <hip/hip_runtime.h>
#include <hip/hip_bf16.h>

#define NU 50000
#define NT 100000
#define NE 600000
#define FD 128
#define LDS_STRIDE 136   // 128 + 8 halfs: 272B rows -> rotated banks, 16B aligned

typedef __attribute__((ext_vector_type(16))) _Float16 v16h;
typedef __attribute__((ext_vector_type(8)))  float    v8f;

// ---------- ordered-float encoding for atomicMax on floats ----------
__device__ __forceinline__ unsigned f2ord(float f) {
    unsigned u = __float_as_uint(f);
    return (u & 0x80000000u) ? ~u : (u | 0x80000000u);
}
__device__ __forceinline__ float ord2f(unsigned u) {
    return (u & 0x80000000u) ? __uint_as_float(u & 0x7fffffffu)
                             : __uint_as_float(~u);
}
#define ORD_NEG_INF 0x007FFFFFu   // f2ord(-inf)

// ---------- weight prep: Wt[o][k] = (f16) W[k][o], 128x128 ----------
__global__ __launch_bounds__(256) void w_prep(const float* __restrict__ W,
                                              _Float16* __restrict__ Wt) {
    int idx = blockIdx.x * 256 + threadIdx.x;      // 0..16383
    int o = idx >> 7, k = idx & 127;
    Wt[idx] = (_Float16)W[k * FD + o];
}

// ---------- WMMA GEMM: Y[N,128] = X[N,128] @ W[128,128] (+opt bias) ----------
// Wt is f16, transposed [out][in]. 256-thread block = 8 waves.
// The 16x128 A tile is staged once into LDS as f16; each wave then owns one
// 16-col tile and runs 4x v_wmma_f32_16x16x32_f16 over K=128.
__global__ __launch_bounds__(256) void gemm128(const float* __restrict__ X,
                                               const _Float16* __restrict__ Wt,
                                               const float* __restrict__ bias,
                                               float* __restrict__ Y) {
    __shared__ _Float16 lds_a[16 * LDS_STRIDE];    // ~4.3 KB

    const int tid  = threadIdx.x;
    const int row0 = blockIdx.x * 16;

    // ---- cooperative stage: X tile (16x128 f32) -> LDS f16 ----
    {
        const int r  = tid >> 4;                   // 0..15
        const int ks = (tid & 15) * 8;             // 0,8,...,120
        const float* xp = X + (size_t)(row0 + r) * FD + ks;
        _Float16 tmp[8];
#pragma unroll
        for (int e = 0; e < 8; ++e) tmp[e] = (_Float16)xp[e];
#pragma unroll
        for (int e = 0; e < 8; ++e) lds_a[r * LDS_STRIDE + ks + e] = tmp[e];
    }
    __syncthreads();

    const int lane  = tid & 31;
    const int colt  = tid >> 5;                    // 0..7
    const int m     = lane & 15;
    const int khalf = lane >> 4;
    const int col   = colt * 16 + m;
    const _Float16* wr = Wt + (size_t)col * FD;
    const _Float16* ar = lds_a + m * LDS_STRIDE;

    v8f c = {};
#pragma unroll
    for (int kb = 0; kb < FD; kb += 32) {
        v16h a, b;
        const int ka = kb + khalf * 8;             // A: K = khalf*8 + e (+16)
#pragma unroll
        for (int e = 0; e < 8; ++e) {
            a[e]     = ar[ka + e];
            a[e + 8] = ar[ka + 16 + e];
        }
        const int kc = kb + khalf * 8;             // B: K = khalf*8 + e (+16)
#pragma unroll
        for (int e = 0; e < 8; ++e) {
            b[e]     = wr[kc + e];
            b[e + 8] = wr[kc + 16 + e];
        }
        c = __builtin_amdgcn_wmma_f32_16x16x32_f16(false, a, false, b,
                                                   (short)0, c, false, false);
    }
    const float bb = bias ? bias[col] : 0.0f;
#pragma unroll
    for (int r = 0; r < 8; ++r)
        Y[(size_t)(row0 + r + 8 * khalf) * FD + col] = c[r] + bb;
}

// ---------- a[n,h] = sum_c H[n, h*32+c] * att[h*32+c] ----------
__global__ __launch_bounds__(256) void att_dot(const float* __restrict__ H,
                                               const float* __restrict__ att,
                                               float* __restrict__ A, int n4) {
    int idx = blockIdx.x * 256 + threadIdx.x;
    if (idx >= n4) return;
    int n = idx >> 2, h = idx & 3;
    const float* hp = H + (size_t)n * FD + h * 32;
    const float* ap = att + h * 32;
    float s = 0.f;
#pragma unroll
    for (int c = 0; c < 32; ++c) s += hp[c] * ap[c];
    A[idx] = s;
}

__global__ __launch_bounds__(256) void fill_u32(unsigned* __restrict__ p,
                                                unsigned v, int n) {
    int i = blockIdx.x * 256 + threadIdx.x;
    if (i < n) p[i] = v;
}

// ---------- edge phase A: e = leaky_relu(a_s[src]+a_d[dst]); seg-max ----------
__global__ __launch_bounds__(256) void edge_logits(const float* __restrict__ As,
                                                   const float* __restrict__ Ad,
                                                   const int* __restrict__ src,
                                                   const int* __restrict__ dst,
                                                   float* __restrict__ ebuf,
                                                   unsigned* __restrict__ mx) {
    int i = blockIdx.x * 256 + threadIdx.x;
    if (i >= NE) return;
    int s = src[i], d = dst[i];
#pragma unroll
    for (int h = 0; h < 4; ++h) {
        float v = As[s * 4 + h] + Ad[d * 4 + h];
        v = (v > 0.f) ? v : 0.2f * v;
        ebuf[(size_t)i * 4 + h] = v;
        atomicMax(&mx[d * 4 + h], f2ord(v));
    }
}

// ---------- edge phase B: ex = exp(e - m[dst]); den[dst] += ex ----------
__global__ __launch_bounds__(256) void edge_exp(const int* __restrict__ dst,
                                                float* __restrict__ ebuf,
                                                const unsigned* __restrict__ mx,
                                                float* __restrict__ den) {
    int i = blockIdx.x * 256 + threadIdx.x;
    if (i >= NE) return;
    int d = dst[i];
#pragma unroll
    for (int h = 0; h < 4; ++h) {
        float m = ord2f(mx[d * 4 + h]);
        if (!(m > -3.0e38f)) m = 0.0f;             // empty-segment guard
        float ex = __expf(ebuf[(size_t)i * 4 + h] - m);
        ebuf[(size_t)i * 4 + h] = ex;
        atomicAdd(&den[d * 4 + h], ex);
    }
}

// ---------- edge phase C: acc[dst, f] += alpha * Hs[src, f] ----------
__global__ __launch_bounds__(256) void edge_scatter(const int* __restrict__ src,
                                                    const int* __restrict__ dst,
                                                    const float* __restrict__ ebuf,
                                                    const float* __restrict__ den,
                                                    const float* __restrict__ Hs,
                                                    float* __restrict__ acc) {
    int idx = blockIdx.x * 256 + threadIdx.x;      // NE*128 = 76.8M < 2^31
    int e = idx >> 7, f = idx & 127, h = f >> 5;
    int s = src[e], d = dst[e];
    float alpha = ebuf[(size_t)e * 4 + h] / (den[d * 4 + h] + 1e-16f);
    atomicAdd(&acc[(size_t)d * FD + f], alpha * Hs[(size_t)s * FD + f]);
}

// ---------- out = elu(acc + bias) ----------
__global__ __launch_bounds__(256) void bias_elu(const float* __restrict__ acc,
                                                const float* __restrict__ bias,
                                                float* __restrict__ out, int n) {
    int idx = blockIdx.x * 256 + threadIdx.x;
    if (idx >= n) return;
    float v = acc[idx] + bias[idx & 127];
    out[idx] = (v > 0.f) ? v : (__expf(v) - 1.0f);
}

// ---------- final: out[n,3] = xt[n,:] @ W[128,3] + b ----------
__global__ __launch_bounds__(256) void final_lin(const float* __restrict__ X,
                                                 const float* __restrict__ W,
                                                 const float* __restrict__ b,
                                                 float* __restrict__ out) {
    int n = blockIdx.x * 256 + threadIdx.x;
    if (n >= NT) return;
    const float* xr = X + (size_t)n * FD;
    float s0 = b[0], s1 = b[1], s2 = b[2];
#pragma unroll 4
    for (int k = 0; k < FD; ++k) {
        float x = xr[k];
        s0 += x * W[k * 3 + 0];
        s1 += x * W[k * 3 + 1];
        s2 += x * W[k * 3 + 2];
    }
    out[n * 3 + 0] = s0; out[n * 3 + 1] = s1; out[n * 3 + 2] = s2;
}

// =====================================================================
extern "C" void kernel_launch(void* const* d_in, const int* in_sizes, int n_in,
                              void* d_out, int out_size, void* d_ws, size_t ws_size,
                              hipStream_t stream) {
    const float* x_user = (const float*)d_in[0];
    const float* x_tweet = (const float*)d_in[1];
    const int* e_src[2] = { (const int*)d_in[2], (const int*)d_in[4] }; // ut, tu
    const int* e_dst[2] = { (const int*)d_in[3], (const int*)d_in[5] };

    // params flattened in insertion (dict) order
    const float* Wemb[2] = { (const float*)d_in[6], (const float*)d_in[8] };
    const float* bemb[2] = { (const float*)d_in[7], (const float*)d_in[9] };
    const float* Wlin = (const float*)d_in[10];
    const float* blin = (const float*)d_in[11];
    struct Gat { const float *Wsrc, *Wdst, *atts, *attd, *bias; } gat[4];
    for (int i = 0; i < 4; ++i) {                   // conv0_ut, conv0_tu, conv1_ut, conv1_tu
        int base = 12 + i * 5;
        gat[i] = { (const float*)d_in[base + 0], (const float*)d_in[base + 1],
                   (const float*)d_in[base + 2], (const float*)d_in[base + 3],
                   (const float*)d_in[base + 4] };
    }

    // ---- carve workspace ----
    char* ws = (char*)d_ws; size_t off = 0;
    auto carve = [&](size_t bytes) -> void* {
        void* p = ws + off; off = (off + bytes + 255) & ~(size_t)255; return p;
    };
    float* xu[2]  = { (float*)carve((size_t)NU * FD * 4), (float*)carve((size_t)NU * FD * 4) };
    float* xt[2]  = { (float*)carve((size_t)NT * FD * 4), (float*)carve((size_t)NT * FD * 4) };
    float* hs     = (float*)carve((size_t)NT * FD * 4);     // src projection (max size)
    float* hd     = (float*)carve((size_t)NT * FD * 4);     // dst projection; reused as acc
    float* a_s    = (float*)carve((size_t)NT * 4 * 4);
    float* a_d    = (float*)carve((size_t)NT * 4 * 4);
    float* ebuf   = (float*)carve((size_t)NE * 4 * 4);
    unsigned* mx  = (unsigned*)carve((size_t)NT * 4 * 4);
    float* den    = (float*)carve((size_t)NT * 4 * 4);
    _Float16* w16 = (_Float16*)carve((size_t)10 * FD * FD * 2);
    (void)ws_size; (void)in_sizes; (void)n_in; (void)out_size;

    // ---- weight prep: slots 0,1 = embeddings; 2+2i,3+2i = gat[i] Wsrc/Wdst ----
    const int WGRID = (FD * FD) / 256;
    w_prep<<<WGRID, 256, 0, stream>>>(Wemb[0], w16 + 0 * FD * FD);
    w_prep<<<WGRID, 256, 0, stream>>>(Wemb[1], w16 + 1 * FD * FD);
    for (int i = 0; i < 4; ++i) {
        w_prep<<<WGRID, 256, 0, stream>>>(gat[i].Wsrc, w16 + (2 + 2 * i) * FD * FD);
        w_prep<<<WGRID, 256, 0, stream>>>(gat[i].Wdst, w16 + (3 + 2 * i) * FD * FD);
    }

    // ---- embeddings ----
    gemm128<<<NU / 16, 256, 0, stream>>>(x_user,  w16 + 0 * FD * FD, bemb[0], xu[0]);
    gemm128<<<NT / 16, 256, 0, stream>>>(x_tweet, w16 + 1 * FD * FD, bemb[1], xt[0]);

    const int EB = (NE + 255) / 256;
    int cu = 0, ct = 0;                             // current xu / xt buffers
    for (int l = 0; l < 2; ++l) {
        // both convs of a layer read the OLD xu/xt; write into the other buffer
        for (int dir = 0; dir < 2; ++dir) {         // dir 0: u->t, dir 1: t->u
            const int gi = l * 2 + dir;
            const float* Xs = dir == 0 ? xu[cu] : xt[ct];
            const float* Xd = dir == 0 ? xt[ct] : xu[cu];
            const int Ns = dir == 0 ? NU : NT;
            const int Nd = dir == 0 ? NT : NU;
            float* xnew = dir == 0 ? xt[ct ^ 1] : xu[cu ^ 1];

            gemm128<<<Ns / 16, 256, 0, stream>>>(Xs, w16 + (2 + 2 * gi) * FD * FD, nullptr, hs);
            gemm128<<<Nd / 16, 256, 0, stream>>>(Xd, w16 + (3 + 2 * gi) * FD * FD, nullptr, hd);
            att_dot<<<(Ns * 4 + 255) / 256, 256, 0, stream>>>(hs, gat[gi].atts, a_s, Ns * 4);
            att_dot<<<(Nd * 4 + 255) / 256, 256, 0, stream>>>(hd, gat[gi].attd, a_d, Nd * 4);

            float* acc = hd;                        // hd no longer needed -> reuse as acc
            fill_u32<<<(Nd * 4 + 255) / 256, 256, 0, stream>>>(mx, ORD_NEG_INF, Nd * 4);
            fill_u32<<<(Nd * 4 + 255) / 256, 256, 0, stream>>>((unsigned*)den, 0u, Nd * 4);
            fill_u32<<<(Nd * FD + 255) / 256, 256, 0, stream>>>((unsigned*)acc, 0u, Nd * FD);

            edge_logits<<<EB, 256, 0, stream>>>(a_s, a_d, e_src[dir], e_dst[dir], ebuf, mx);
            edge_exp<<<EB, 256, 0, stream>>>(e_dst[dir], ebuf, mx, den);
            edge_scatter<<<(NE * FD) / 256, 256, 0, stream>>>(e_src[dir], e_dst[dir],
                                                              ebuf, den, hs, acc);
            bias_elu<<<(Nd * FD + 255) / 256, 256, 0, stream>>>(acc, gat[gi].bias, xnew, Nd * FD);
        }
        cu ^= 1; ct ^= 1;
    }

    final_lin<<<(NT + 255) / 256, 256, 0, stream>>>(xt[ct], Wlin, blin, (float*)d_out);
}
